// CudaFastWeightSumTwoLinearTransformerLayer_15891378995458
// MI455X (gfx1250) — matrix operand order repaired and checked by reading, not compiled, not run on hardware
//
#include <hip/hip_runtime.h>

typedef __bf16 bf16;
typedef __attribute__((ext_vector_type(16))) __bf16 v16bf;
typedef __attribute__((ext_vector_type(8)))  __bf16 v8bf;
typedef __attribute__((ext_vector_type(8)))  float   v8f;

#define LL   256
#define BSZ  64
#define NHD  8
#define DHD  64
#define DMD  512
#define EPSL 1e-5f
#define SCL  0.125f   // 1/sqrt(64)

// ---------------------------------------------------------------------------
// WMMA fragment helpers (gfx1250 layouts, wave32)
// A (16x32, 16-bit): lane m=lane&15; lanes 0-15 hold K {0..7,16..23},
//                    lanes 16-31 hold K {8..15,24..31}
// B (32x16, 16-bit): col n=lane&15; K split across lane halves as for A
// C/D (16x16 f32):   VGPR r -> row m = r + 8*(lane>=16), col n = lane&15
// ---------------------------------------------------------------------------
__device__ __forceinline__ v8f zero8() {
  v8f z;
#pragma unroll
  for (int i = 0; i < 8; ++i) z[i] = 0.0f;
  return z;
}

__device__ __forceinline__ v16bf frag_a(const bf16* p, int ld, int lane) {
  const int m  = lane & 15;
  const int ko = (lane < 16) ? 0 : 8;
  v8bf lo = *(const v8bf*)(p + m * ld + ko);
  v8bf hi = *(const v8bf*)(p + m * ld + ko + 16);
  v16bf r;
#pragma unroll
  for (int i = 0; i < 8; ++i) { r[i] = lo[i]; r[i + 8] = hi[i]; }
  return r;
}

__device__ __forceinline__ v16bf frag_b(const bf16* p, int ld, int lane) {
  const int n  = lane & 15;
  const int ko = (lane < 16) ? 0 : 8;
  v16bf r;
#pragma unroll
  for (int i = 0; i < 8; ++i) { r[i] = p[(ko + i) * ld + n]; r[i + 8] = p[(ko + 16 + i) * ld + n]; }
  return r;
}

__device__ __forceinline__ v8f wmma_bf(v16bf a, v16bf b, v8f c) {
  return __builtin_amdgcn_wmma_f32_16x16x32_bf16(false, a, false, b,
                                                 (short)0, c, false, false);
}

// ---------------------------------------------------------------------------
// Kernel 1: convert h -> bf16, pre-transpose weights into K-major bf16
//   wT[d][e]  = qkv_w[e][d]   (512 x 2048)
//   oT[e][dp] = o_w[dp][e]    (512 x 512)
// ---------------------------------------------------------------------------
__global__ __launch_bounds__(256) void cvt_kernel(
    const float* __restrict__ h, const float* __restrict__ qkv_w,
    const float* __restrict__ o_w, bf16* __restrict__ hb,
    bf16* __restrict__ wT, bf16* __restrict__ oT) {
  const size_t i = (size_t)blockIdx.x * 256 + threadIdx.x;
  if (i < (size_t)LL * BSZ * DMD) hb[i] = (bf16)h[i];
  if (i < (size_t)DMD * 2048) {
    const int d = (int)(i >> 11), e = (int)(i & 2047);
    wT[i] = (bf16)qkv_w[(size_t)e * DMD + d];
  }
  if (i < (size_t)DMD * DMD) {
    const int e = (int)(i >> 9), dp = (int)(i & 511);
    oT[i] = (bf16)o_w[(size_t)dp * DMD + e];
  }
}

// ---------------------------------------------------------------------------
// Kernel 2: qkvb[16384 x 2048] = hb[16384 x 512] @ wT  (bf16 WMMA, f32 accum)
// Block = 128x128 tile, 8 waves (4 m x 2 n), each wave 32x64 (2x4 WMMA tiles)
// K-loop is software-pipelined (next fragments loaded before current WMMAs)
// and unrolled x2 so the double-buffer rotation becomes register renaming
// instead of v_mov chains.
// ---------------------------------------------------------------------------
__global__ __launch_bounds__(256) void gemm_qkv_kernel(
    const bf16* __restrict__ hb, const bf16* __restrict__ wT,
    bf16* __restrict__ qkvb) {
  const int lane = threadIdx.x & 31;
  const int wv   = threadIdx.x >> 5;
  const int wm   = wv >> 1, wn = wv & 1;
  const int rm0  = blockIdx.y * 128 + wm * 32;
  const int cn0  = blockIdx.x * 128 + wn * 64;

  v8f acc[2][4];
#pragma unroll
  for (int i = 0; i < 2; ++i)
#pragma unroll
    for (int j = 0; j < 4; ++j) acc[i][j] = zero8();

  // prologue: fragments for kk = 0
  v16bf a0 = frag_a(hb + (size_t)(rm0 + 0)  * DMD, DMD, lane);
  v16bf a1 = frag_a(hb + (size_t)(rm0 + 16) * DMD, DMD, lane);
  v16bf bf[4];
#pragma unroll
  for (int j = 0; j < 4; ++j)
    bf[j] = frag_b(wT + cn0 + 16 * j, 2048, lane);

#pragma unroll 2
  for (int kk = 0; kk < DMD; kk += 32) {
    // issue next iteration's loads first (last iter reloads kk harmlessly)
    const int kn = (kk + 32 < DMD) ? kk + 32 : kk;
    __builtin_prefetch(hb + (size_t)rm0 * DMD + kn, 0, 1);
    v16bf na0 = frag_a(hb + (size_t)(rm0 + 0)  * DMD + kn, DMD, lane);
    v16bf na1 = frag_a(hb + (size_t)(rm0 + 16) * DMD + kn, DMD, lane);
    v16bf nb[4];
#pragma unroll
    for (int j = 0; j < 4; ++j)
      nb[j] = frag_b(wT + (size_t)kn * 2048 + cn0 + 16 * j, 2048, lane);

    // compute with current fragments while next loads are in flight
#pragma unroll
    for (int j = 0; j < 4; ++j) {
      acc[0][j] = wmma_bf(a0, bf[j], acc[0][j]);
      acc[1][j] = wmma_bf(a1, bf[j], acc[1][j]);
    }
    a0 = na0; a1 = na1;
#pragma unroll
    for (int j = 0; j < 4; ++j) bf[j] = nb[j];
  }

  const int n = lane & 15, hf = lane >> 4;
#pragma unroll
  for (int i = 0; i < 2; ++i)
#pragma unroll
    for (int j = 0; j < 4; ++j)
#pragma unroll
      for (int r = 0; r < 8; ++r) {
        const int row = rm0 + 16 * i + r + 8 * hf;
        const int col = cn0 + 16 * j + n;
        qkvb[(size_t)row * 2048 + col] = (bf16)acc[i][j][r];
      }
}

// ---------------------------------------------------------------------------
// Kernel 3: phi normalization + key mixing.
// One wave per (l,b,head); lane handles d and d+32.
//   Qb[b,h,l,d] (row-major), Vb[b,h,l,d], KbT[b,h,d,l] (transposed for WMMA-B)
// ---------------------------------------------------------------------------
__device__ __forceinline__ float phi1(float x) {
  return x > 0.0f ? x + 1.0f : __expf(x);   // elu(x)+1
}
__device__ __forceinline__ float wsum(float v) {
#pragma unroll
  for (int off = 1; off < 32; off <<= 1) v += __shfl_xor(v, off, 32);
  return v;
}

__global__ __launch_bounds__(256) void phi_kernel(
    const bf16* __restrict__ qkvb, const float* __restrict__ pi0,
    const float* __restrict__ pi1, bf16* __restrict__ Qb,
    bf16* __restrict__ KbT, bf16* __restrict__ Vb) {
  const int lane = threadIdx.x & 31;
  const int wv   = threadIdx.x >> 5;
  const int unit = blockIdx.x * 8 + wv;        // 0 .. L*B*NH-1
  const int lb   = unit >> 3;
  const int hh   = unit & 7;
  const int l    = lb >> 6;                    // lb = l*B + b, B=64
  const int b    = lb & 63;
  const bf16* base = qkvb + (size_t)lb * 2048 + hh * 256;
  const int d0 = lane, d1 = lane + 32;

  float q0 = phi1((float)base[d0]),        q1 = phi1((float)base[d1]);
  float a0 = phi1((float)base[64 + d0]),   a1 = phi1((float)base[64 + d1]);
  float c0 = phi1((float)base[128 + d0]),  c1 = phi1((float)base[128 + d1]);
  float v0 = (float)base[192 + d0],        v1 = (float)base[192 + d1];

  const float qs = wsum(q0 + q1);
  const float as = wsum(a0 + a1);
  const float cs = wsum(c0 + c1);
  q0 /= qs; q1 /= qs; a0 /= as; a1 /= as; c0 /= cs; c1 /= cs;

  const float p0 = fminf(fmaxf(pi0[hh * 256 + l], 0.0f), 1.0f);
  const float p1 = fminf(fmaxf(pi1[hh * 256 + l], 0.0f), 1.0f);
  const float k0 = a0 * p0 + c0 * p1;      // SCALE_W == 1
  const float k1 = a1 * p0 + c1 * p1;

  const size_t bh = (size_t)b * NHD + hh;
  Qb[(bh * LL + l) * DHD + d0]  = (bf16)q0;
  Qb[(bh * LL + l) * DHD + d1]  = (bf16)q1;
  Vb[(bh * LL + l) * DHD + d0]  = (bf16)v0;
  Vb[(bh * LL + l) * DHD + d1]  = (bf16)v1;
  KbT[(bh * DHD + d0) * LL + l] = (bf16)k0;
  KbT[(bh * DHD + d1) * LL + l] = (bf16)k1;
}

// ---------------------------------------------------------------------------
// Kernel 4: causal linear attention per (b,h):
//   S = mask(Q K^T); denom = rowsum(S); O = S V; out = SCL*O/(denom+EPS)
// One block per (b,h); 8 waves; each wave owns t-tiles {w, w+8} (16 rows).
// S computed with WMMA, masked + row-summed in C layout, staged to LDS as
// bf16 to become the A operand of the O-accumulating WMMA.
// ---------------------------------------------------------------------------
__global__ __launch_bounds__(256) void attn_kernel(
    const bf16* __restrict__ Qb, const bf16* __restrict__ KbT,
    const bf16* __restrict__ Vb, bf16* __restrict__ outb) {
  __shared__ __align__(16) bf16 sstage[8][16][40];   // per-wave 16x32 (+pad)
  const int lane = threadIdx.x & 31;
  const int wv   = threadIdx.x >> 5;
  const int bh   = blockIdx.x;
  const int b    = bh >> 3, hh = bh & 7;
  const bf16* Qh = Qb  + (size_t)bh * LL * DHD;
  const bf16* Kh = KbT + (size_t)bh * DHD * LL;
  const bf16* Vh = Vb  + (size_t)bh * LL * DHD;
  const int n = lane & 15, hf = lane >> 4;

  for (int tt = wv; tt < 16; tt += 8) {
    const int t0 = tt * 16;
    const v16bf qf0 = frag_a(Qh + (size_t)t0 * DHD + 0,  DHD, lane);
    const v16bf qf1 = frag_a(Qh + (size_t)t0 * DHD + 32, DHD, lane);

    v8f o[4];
#pragma unroll
    for (int j = 0; j < 4; ++j) o[j] = zero8();
    float den[8];
#pragma unroll
    for (int r = 0; r < 8; ++r) den[r] = 0.0f;

    for (int s0 = 0; s0 <= t0; s0 += 32) {
#pragma unroll
      for (int jj = 0; jj < 2; ++jj) {
        const int ss = s0 + 16 * jj;
        if (ss <= t0) {
          v8f sacc = zero8();
          sacc = wmma_bf(qf0, frag_b(Kh + ss, LL, lane), sacc);
          sacc = wmma_bf(qf1, frag_b(Kh + (size_t)32 * LL + ss, LL, lane), sacc);
          if (ss == t0) {                       // diagonal tile: mask s > t
#pragma unroll
            for (int r = 0; r < 8; ++r) {
              const int tg = t0 + r + 8 * hf, sg = ss + n;
              if (sg > tg) sacc[r] = 0.0f;
            }
          }
#pragma unroll
          for (int r = 0; r < 8; ++r) {
            den[r] += sacc[r];                  // denom = rowsum of masked S
            sstage[wv][r + 8 * hf][jj * 16 + n] = (bf16)sacc[r];
          }
        } else {
#pragma unroll
          for (int r = 0; r < 8; ++r)
            sstage[wv][r + 8 * hf][jj * 16 + n] = (bf16)0.0f;
        }
      }
      const v16bf sa = frag_a(&sstage[wv][0][0], 40, lane);
#pragma unroll
      for (int j = 0; j < 4; ++j) {
        const v16bf vb = frag_b(Vh + (size_t)s0 * DHD + 16 * j, DHD, lane);
        o[j] = wmma_bf(sa, vb, o[j]);
      }
    }

    // reduce denom across the 16 lanes sharing each row, build scale factor
#pragma unroll
    for (int r = 0; r < 8; ++r) {
      float d = den[r];
#pragma unroll
      for (int off = 1; off < 16; off <<= 1) d += __shfl_xor(d, off, 32);
      den[r] = SCL / (d + EPSL);
    }

    // store to [L, B, NH*DH] bf16 for the output GEMM
#pragma unroll
    for (int j = 0; j < 4; ++j)
#pragma unroll
      for (int r = 0; r < 8; ++r) {
        const int l   = t0 + r + 8 * hf;
        const int col = hh * DHD + 16 * j + n;
        outb[((size_t)l * BSZ + b) * DMD + col] = (bf16)(o[j][r] * den[r]);
      }
  }
}

// ---------------------------------------------------------------------------
// Kernel 5: x = h + outb @ oT ; LayerNorm(x) -> d_out  (fused)
// Block = 64 rows; wave w owns cols [64w, 64w+64): 4x4 WMMA tiles.
// K-loop software-pipelined + unrolled x2 like gemm_qkv.
// LayerNorm via register partials -> small LDS -> normalize from registers.
// ---------------------------------------------------------------------------
__global__ __launch_bounds__(256) void gemm_out_ln_kernel(
    const bf16* __restrict__ outb, const bf16* __restrict__ oT,
    const float* __restrict__ h, const float* __restrict__ ln_g,
    const float* __restrict__ ln_b, float* __restrict__ out) {
  __shared__ float partial[8][64][2];
  __shared__ float muvar[64][2];
  const int lane = threadIdx.x & 31;
  const int wv   = threadIdx.x >> 5;
  const int r0   = blockIdx.x * 64;
  const int c0   = wv * 64;
  const int n = lane & 15, hf = lane >> 4;

  v8f acc[4][4];
#pragma unroll
  for (int i = 0; i < 4; ++i)
#pragma unroll
    for (int j = 0; j < 4; ++j) acc[i][j] = zero8();

  // prologue: fragments for kk = 0
  v16bf a[4], bb[4];
#pragma unroll
  for (int i = 0; i < 4; ++i)
    a[i] = frag_a(outb + (size_t)(r0 + 16 * i) * DMD, DMD, lane);
#pragma unroll
  for (int j = 0; j < 4; ++j)
    bb[j] = frag_b(oT + c0 + 16 * j, DMD, lane);

#pragma unroll 2
  for (int kk = 0; kk < DMD; kk += 32) {
    const int kn = (kk + 32 < DMD) ? kk + 32 : kk;
    __builtin_prefetch(outb + (size_t)r0 * DMD + kn, 0, 1);
    v16bf na[4], nb[4];
#pragma unroll
    for (int i = 0; i < 4; ++i)
      na[i] = frag_a(outb + (size_t)(r0 + 16 * i) * DMD + kn, DMD, lane);
#pragma unroll
    for (int j = 0; j < 4; ++j)
      nb[j] = frag_b(oT + (size_t)kn * DMD + c0 + 16 * j, DMD, lane);

#pragma unroll
    for (int i = 0; i < 4; ++i)
#pragma unroll
      for (int j = 0; j < 4; ++j) acc[i][j] = wmma_bf(a[i], bb[j], acc[i][j]);

#pragma unroll
    for (int i = 0; i < 4; ++i) a[i] = na[i];
#pragma unroll
    for (int j = 0; j < 4; ++j) bb[j] = nb[j];
  }

  // residual add (keep x in the accumulators)
#pragma unroll
  for (int i = 0; i < 4; ++i)
#pragma unroll
    for (int j = 0; j < 4; ++j)
#pragma unroll
      for (int r = 0; r < 8; ++r) {
        const int lr  = 16 * i + r + 8 * hf;
        const int col = c0 + 16 * j + n;
        acc[i][j][r] += h[(size_t)(r0 + lr) * DMD + col];
      }

  // per-wave row partial sums / sumsq over this wave's 64 cols
#pragma unroll
  for (int i = 0; i < 4; ++i)
#pragma unroll
    for (int r = 0; r < 8; ++r) {
      float s1 = 0.0f, s2 = 0.0f;
#pragma unroll
      for (int j = 0; j < 4; ++j) {
        const float x = acc[i][j][r];
        s1 += x; s2 += x * x;
      }
#pragma unroll
      for (int off = 1; off < 16; off <<= 1) {
        s1 += __shfl_xor(s1, off, 32);
        s2 += __shfl_xor(s2, off, 32);
      }
      if (n == 0) {
        const int lr = 16 * i + r + 8 * hf;
        partial[wv][lr][0] = s1;
        partial[wv][lr][1] = s2;
      }
    }
  __syncthreads();

  if (threadIdx.x < 64) {
    float s1 = 0.0f, s2 = 0.0f;
#pragma unroll
    for (int ww = 0; ww < 8; ++ww) {
      s1 += partial[ww][threadIdx.x][0];
      s2 += partial[ww][threadIdx.x][1];
    }
    const float mu  = s1 * (1.0f / DMD);
    const float var = s2 * (1.0f / DMD) - mu * mu;
    muvar[threadIdx.x][0] = mu;
    muvar[threadIdx.x][1] = rsqrtf(var + EPSL);
  }
  __syncthreads();

#pragma unroll
  for (int i = 0; i < 4; ++i)
#pragma unroll
    for (int j = 0; j < 4; ++j)
#pragma unroll
      for (int r = 0; r < 8; ++r) {
        const int lr  = 16 * i + r + 8 * hf;
        const int col = c0 + 16 * j + n;
        const float x = (acc[i][j][r] - muvar[lr][0]) * muvar[lr][1];
        out[(size_t)(r0 + lr) * DMD + col] = x * ln_g[col] + ln_b[col];
      }
}

// ---------------------------------------------------------------------------
extern "C" void kernel_launch(void* const* d_in, const int* in_sizes, int n_in,
                              void* d_out, int out_size, void* d_ws,
                              size_t ws_size, hipStream_t stream) {
  (void)in_sizes; (void)n_in; (void)out_size; (void)ws_size;
  const float* h     = (const float*)d_in[0];
  const float* qkv_w = (const float*)d_in[1];
  const float* o_w   = (const float*)d_in[2];
  const float* pi0   = (const float*)d_in[3];
  const float* pi1   = (const float*)d_in[4];
  const float* ln_g  = (const float*)d_in[5];
  const float* ln_b  = (const float*)d_in[6];
  float* out = (float*)d_out;

  char* ws = (char*)d_ws;                         // ~147 MB total
  bf16* hb   = (bf16*)ws; ws += (size_t)16384 * 512 * 2;   // 16 MB
  bf16* wT   = (bf16*)ws; ws += (size_t)512 * 2048 * 2;    //  2 MB
  bf16* oT   = (bf16*)ws; ws += (size_t)512 * 512 * 2;     // 0.5 MB
  bf16* qkvb = (bf16*)ws; ws += (size_t)16384 * 2048 * 2;  // 64 MB
  bf16* Qb   = (bf16*)ws; ws += (size_t)512 * 256 * 64 * 2;// 16 MB
  bf16* KbT  = (bf16*)ws; ws += (size_t)512 * 64 * 256 * 2;// 16 MB
  bf16* Vb   = (bf16*)ws; ws += (size_t)512 * 256 * 64 * 2;// 16 MB
  bf16* outb = (bf16*)ws; ws += (size_t)16384 * 512 * 2;   // 16 MB

  cvt_kernel<<<32768, 256, 0, stream>>>(h, qkv_w, o_w, hb, wT, oT);
  gemm_qkv_kernel<<<dim3(16, 128), 256, 0, stream>>>(hb, wT, qkvb);
  phi_kernel<<<16384, 256, 0, stream>>>(qkvb, pi0, pi1, Qb, KbT, Vb);
  attn_kernel<<<512, 256, 0, stream>>>(Qb, KbT, Vb, outb);
  gemm_out_ln_kernel<<<256, 256, 0, stream>>>(outb, oT, h, ln_g, ln_b, out);
}
